// BlockSparseMoE_15882789061249
// MI455X (gfx1250) — matrix operand order repaired and checked by reading, not compiled, not run on hardware
//
#include <hip/hip_runtime.h>
#include <hip/hip_bf16.h>
#include <math.h>

// Problem constants (match reference setup_inputs)
#define T_TOK 1024
#define H_DIM 2048
#define F_DIM 1408
#define N_EXP 16
#define TOPK  6

typedef __attribute__((ext_vector_type(16))) __bf16 v16bf;
typedef __attribute__((ext_vector_type(8)))  __bf16 v8bf;
typedef __attribute__((ext_vector_type(8)))  float  v8f;

// ---------------------------------------------------------------------------
// Fragment loaders (contiguous k = g*16 + i order in both A and B, wave32)
// A (16x32 bf16, row-major, K contiguous): lane m = lane&15, half g = lane>>4
//   frag[0..7]  = A[m][k0 + g*16 + 0..7]
//   frag[8..15] = A[m][k0 + g*16 + 8 + 0..7]
// ---------------------------------------------------------------------------
__device__ __forceinline__ v16bf load_a_frag_bf16(const __bf16* __restrict__ A,
                                                  int lda, int row, int g, int k0) {
    const __bf16* r = A + (size_t)row * lda + k0 + g * 16;
    const v8bf lo = *(const v8bf*)(r);
    const v8bf hi = *(const v8bf*)(r + 8);
    v16bf f;
#pragma unroll
    for (int i = 0; i < 8; ++i) { f[i] = lo[i]; f[i + 8] = hi[i]; }
    return f;
}

// B (32x16 bf16, from fp32 weights stored [N, K] with K contiguous):
// lane n = lane&15, half g = lane>>4; frag[i] = W[n][k0 + g*16 + i], i=0..15.
// p must already point at W + n*ldb + k0 + g*16 (64B-aligned here).
__device__ __forceinline__ v16bf load_b_frag_f32(const float* __restrict__ p) {
    const float4 q0 = ((const float4*)p)[0];
    const float4 q1 = ((const float4*)p)[1];
    const float4 q2 = ((const float4*)p)[2];
    const float4 q3 = ((const float4*)p)[3];
    v16bf f;
    f[0]  = (__bf16)q0.x; f[1]  = (__bf16)q0.y; f[2]  = (__bf16)q0.z; f[3]  = (__bf16)q0.w;
    f[4]  = (__bf16)q1.x; f[5]  = (__bf16)q1.y; f[6]  = (__bf16)q1.z; f[7]  = (__bf16)q1.w;
    f[8]  = (__bf16)q2.x; f[9]  = (__bf16)q2.y; f[10] = (__bf16)q2.z; f[11] = (__bf16)q2.w;
    f[12] = (__bf16)q3.x; f[13] = (__bf16)q3.y; f[14] = (__bf16)q3.z; f[15] = (__bf16)q3.w;
    return f;
}

__device__ __forceinline__ v8f wmma_bf16(v16bf a, v16bf b, v8f c) {
    // (neg_a, A, neg_b, B, c_mod, C, reuse_a, reuse_b)
    return __builtin_amdgcn_wmma_f32_16x16x32_bf16(false, a, false, b, (short)0, c,
                                                   false, false);
}

// Fast SiLU: x * rcp(1+exp(-x)) using v_rcp_f32 instead of IEEE divide chains.
__device__ __forceinline__ float fast_silu(float x) {
    return x * __builtin_amdgcn_rcpf(1.0f + __expf(-x));
}

// ---------------------------------------------------------------------------
// Kernel 0: x fp32 -> bf16 (A operand precision for WMMA)
// ---------------------------------------------------------------------------
__global__ void cvt_f32_to_bf16(const float* __restrict__ in,
                                __bf16* __restrict__ out, int n) {
    int i = blockIdx.x * blockDim.x + threadIdx.x;
    const int stride = gridDim.x * blockDim.x;
    for (; i < n; i += stride) out[i] = (__bf16)in[i];
}

// ---------------------------------------------------------------------------
// Kernel 1: router. One wave per token: lane e<16 computes logit_e, lane 0 does
// softmax -> top-6 -> renormalize -> dense route row (zeros off top-k).
// ---------------------------------------------------------------------------
__global__ void router_kernel(const float* __restrict__ x,
                              const float* __restrict__ gw,
                              float* __restrict__ route) {
    const int t = blockIdx.x;
    const int lane = threadIdx.x;
    __shared__ float logits[N_EXP];
    if (lane < N_EXP) {
        const float* xr = x + (size_t)t * H_DIM;
        const float* gr = gw + (size_t)lane * H_DIM;
        float s = 0.f;
        for (int h = 0; h < H_DIM; ++h) s += xr[h] * gr[h];
        logits[lane] = s;
    }
    __syncthreads();
    if (lane == 0) {
        float p[N_EXP], sel[N_EXP];
        float mx = logits[0];
        for (int i = 1; i < N_EXP; ++i) mx = fmaxf(mx, logits[i]);
        float sum = 0.f;
        for (int i = 0; i < N_EXP; ++i) { p[i] = __expf(logits[i] - mx); sum += p[i]; }
        const float rs = __builtin_amdgcn_rcpf(sum);
        for (int i = 0; i < N_EXP; ++i) { p[i] *= rs; sel[i] = 0.f; }
        float ssum = 0.f;
        for (int k = 0; k < TOPK; ++k) {
            int bj = 0; float bv = -1.f;
            for (int j = 0; j < N_EXP; ++j) if (p[j] > bv) { bv = p[j]; bj = j; }
            sel[bj] = bv; ssum += bv; p[bj] = -2.f;
        }
        const float rss = __builtin_amdgcn_rcpf(ssum);
        for (int j = 0; j < N_EXP; ++j) route[(size_t)t * N_EXP + j] = sel[j] * rss;
    }
}

// ---------------------------------------------------------------------------
// Kernel 1b: per-64-token-tile expert-active bitmask (block-sparse metadata).
// mask[tile] bit e == 1  <=>  some token in the tile routes to expert e.
// Both GEMM passes test the SAME bit, so skipped act regions are never read.
// ---------------------------------------------------------------------------
__global__ void route_mask_kernel(const float* __restrict__ route,
                                  unsigned int* __restrict__ mask) {
    const int t0 = blockIdx.x * 64;
    const int tl = threadIdx.x;           // 64 threads, one token each
    __shared__ unsigned int sm;
    if (tl == 0) sm = 0u;
    __syncthreads();
    unsigned int bits = 0u;
    const float* r = route + (size_t)(t0 + tl) * N_EXP;
    for (int e = 0; e < N_EXP; ++e) bits |= (r[e] != 0.0f) ? (1u << e) : 0u;
    atomicOr(&sm, bits);                  // ds_or_b32
    __syncthreads();
    if (tl == 0) mask[blockIdx.x] = sm;
}

// ---------------------------------------------------------------------------
// Kernel 2: gate/up GEMM + SiLU + routing weight -> act[E,T,F] bf16.
// grid = (F/64, T/64, E); block = 128 (4 waves). Each wave owns a 16-wide f
// slab and FOUR 16-token slabs (8 accumulators): each B fragment feeds 4
// WMMAs. Expert is blockIdx.z (slowest) so wv1[e] (23 MB) stays L2-resident.
// ---------------------------------------------------------------------------
__global__ __launch_bounds__(128)
void moe_gate_up(const float* __restrict__ wv1, const __bf16* __restrict__ xb,
                 const float* __restrict__ route,
                 const unsigned int* __restrict__ mask,
                 __bf16* __restrict__ act) {
    const int e  = blockIdx.z;
    const int t0 = blockIdx.y * 64;

    // Block-sparse skip: scalar bit test, uniform across the whole block.
    if (!((mask[blockIdx.y] >> e) & 1u)) return;

    const int wave = threadIdx.x >> 5;
    const int lane = threadIdx.x & 31;
    const int f0w  = blockIdx.x * 64 + wave * 16;
    const int m = lane & 15, g = lane >> 4, n = lane & 15;

    v8f accG[4] = {{}, {}, {}, {}};
    v8f accU[4] = {{}, {}, {}, {}};

    const float* bg_base = wv1 + ((size_t)e * 2 * F_DIM + f0w + n) * H_DIM + g * 16;
    const float* bu_base = wv1 + ((size_t)e * 2 * F_DIM + F_DIM + f0w + n) * H_DIM + g * 16;

    for (int k0 = 0; k0 < H_DIM; k0 += 32) {
        __builtin_prefetch(bg_base + k0 + 64, 0, 0);
        __builtin_prefetch(bu_base + k0 + 64, 0, 0);
        v16bf a[4];
#pragma unroll
        for (int s = 0; s < 4; ++s)
            a[s] = load_a_frag_bf16(xb, H_DIM, t0 + 16 * s + m, g, k0);
        const v16bf bg = load_b_frag_f32(bg_base + k0);
        const v16bf bu = load_b_frag_f32(bu_base + k0);
#pragma unroll
        for (int s = 0; s < 4; ++s) {
            accG[s] = wmma_bf16(a[s], bg, accG[s]);
            accU[s] = wmma_bf16(a[s], bu, accU[s]);
        }
    }

    // Epilogue: act = silu(g)*u * route[t,e]  (routing weight folded in here so
    // the down projection is a plain accumulate over experts).
    const int mg = g * 8;
#pragma unroll
    for (int s = 0; s < 4; ++s) {
#pragma unroll
        for (int r = 0; r < 8; ++r) {
            const int t = t0 + 16 * s + mg + r;
            const float w  = route[(size_t)t * N_EXP + e];
            const float av = fast_silu(accG[s][r]) * accU[s][r] * w;
            act[((size_t)e * T_TOK + t) * F_DIM + f0w + n] = (__bf16)av;
        }
    }
}

// ---------------------------------------------------------------------------
// Kernel 3: down projection + expert combine.
// out[t,h] = sum_e sum_f act[e,t,f] * w2[e,h,f]   (route already in act)
// grid = (H/64, T/64); block = 128. 4 token slabs per wave; accumulates all
// active experts in-register (fused combine).
// ---------------------------------------------------------------------------
__global__ __launch_bounds__(128)
void moe_down(const float* __restrict__ w2, const __bf16* __restrict__ act,
              const unsigned int* __restrict__ mask, float* __restrict__ out) {
    const int t0   = blockIdx.y * 64;
    const int wave = threadIdx.x >> 5;
    const int lane = threadIdx.x & 31;
    const int h0w  = blockIdx.x * 64 + wave * 16;
    const int m = lane & 15, g = lane >> 4, n = lane & 15;

    const unsigned int maskv = mask[blockIdx.y];
    v8f acc[4] = {{}, {}, {}, {}};

    for (int e = 0; e < N_EXP; ++e) {
        if (!((maskv >> e) & 1u)) continue;   // uniform scalar skip

        const __bf16* a_base = act + (size_t)e * T_TOK * F_DIM;
        const float*  b_base = w2 + ((size_t)e * H_DIM + h0w + n) * F_DIM + g * 16;

        for (int k0 = 0; k0 < F_DIM; k0 += 32) {
            __builtin_prefetch(b_base + k0 + 64, 0, 0);
            v16bf a[4];
#pragma unroll
            for (int s = 0; s < 4; ++s)
                a[s] = load_a_frag_bf16(a_base, F_DIM, t0 + 16 * s + m, g, k0);
            const v16bf b = load_b_frag_f32(b_base + k0);
#pragma unroll
            for (int s = 0; s < 4; ++s)
                acc[s] = wmma_bf16(a[s], b, acc[s]);
        }
    }

    const int mg = g * 8;
#pragma unroll
    for (int s = 0; s < 4; ++s) {
#pragma unroll
        for (int r = 0; r < 8; ++r)
            out[(size_t)(t0 + 16 * s + mg + r) * H_DIM + h0w + n] = acc[s][r];
    }
}

// ---------------------------------------------------------------------------
// Host launcher
// ---------------------------------------------------------------------------
extern "C" void kernel_launch(void* const* d_in, const int* in_sizes, int n_in,
                              void* d_out, int out_size, void* d_ws, size_t ws_size,
                              hipStream_t stream) {
    const float* x      = (const float*)d_in[0];   // [T,H]
    const float* gate_w = (const float*)d_in[1];   // [E,H]
    const float* wv1    = (const float*)d_in[2];   // [E,2F,H]
    const float* w2     = (const float*)d_in[3];   // [E,H,F]
    // d_in[4] = top_k (==6, baked into TOPK)
    (void)in_sizes; (void)n_in; (void)out_size; (void)ws_size;

    // Workspace layout (~50.4 MB total):
    //   [0)                     x_bf16 : T*H*2    = 4 MiB
    //   [+4 MiB)                route  : T*E*4    = 64 KiB
    //   [+4 MiB+64 KiB)         mask   : (T/64)*4 = 64 B   (padded to 256 B)
    //   [+4 MiB+64 KiB+256 B)   act    : E*T*F*2  = 44 MiB
    char* ws = (char*)d_ws;
    const size_t off_route = (size_t)T_TOK * H_DIM * 2;
    const size_t off_mask  = off_route + (size_t)T_TOK * N_EXP * 4;
    const size_t off_act   = off_mask + 256;

    __bf16*       xb    = (__bf16*)ws;
    float*        route = (float*)(ws + off_route);
    unsigned int* mask  = (unsigned int*)(ws + off_mask);
    __bf16*       act   = (__bf16*)(ws + off_act);

    cvt_f32_to_bf16<<<2048, 256, 0, stream>>>(x, xb, T_TOK * H_DIM);
    router_kernel<<<T_TOK, 32, 0, stream>>>(x, gate_w, route);
    route_mask_kernel<<<T_TOK / 64, 64, 0, stream>>>(route, mask);
    moe_gate_up<<<dim3(F_DIM / 64, T_TOK / 64, N_EXP), 128, 0, stream>>>(
        wv1, xb, route, mask, act);
    moe_down<<<dim3(H_DIM / 64, T_TOK / 64), 128, 0, stream>>>(
        w2, act, mask, (float*)d_out);
}
